// RNN_66769561584141
// MI455X (gfx1250) — matrix operand order repaired and checked by reading, not compiled, not run on hardware
//
#include <hip/hip_runtime.h>
#include <hip/hip_bf16.h>
#include <math.h>

// ---------------------------------------------------------------------------
// Persistent fused RNN scan for MI455X (gfx1250, wave32, WMMA).
// Block b owns batch rows [16b,16b+16); their hidden state never crosses
// workgroups -> ONE persistent kernel loops all 512 timesteps with hs
// resident in LDS (bf16), then runs the decode head in place.
//  * v_wmma_f32_16x16x32_bf16, f32 accumulate, bias folded into WMMA C.
//  * Weights pre-packed (bf16) to per-lane B-fragment layout: every fragment
//    is 2 coalesced global_load_b128 at IMMEDIATE offsets.
//  * An opaque zero OFFSET (not a laundered pointer!) is added to the weight
//    bases each t-iteration: defeats LICM (no scratch spilling, round 3)
//    while keeping global address-space inference (no flat_load + DScnt
//    coupling, round 4).
//  * B fragments ping-pong double-buffered; next timestep's features
//    prefetched into registers during the GEMMs.
// ---------------------------------------------------------------------------

typedef __attribute__((ext_vector_type(16))) __bf16 v16bf;
typedef __attribute__((ext_vector_type(8)))  float  v8f;

#define B_   1024
#define T_   512
#define F_   256     // IN_FEATS
#define H_   512     // HIDDEN
#define O_   64      // OUT_FEATS
#define ROWS 16      // batch rows per workgroup
#define XK   768     // F_ + H_  (layer-1 K)
#define XSTRIDE 776  // 768 + 8 bf16 pad -> conflict-free ds_read_b128
#define HSTRIDE 520  // 512 + 8 bf16 pad
#define NTHREADS 512 // 16 waves -> 4 waves per SIMD32
#define NT      2    // N-tiles (16 cols) per wave: 16 waves * 2 * 16 = 512

static __device__ __forceinline__ unsigned short f2bf(float f) {
    unsigned u = __builtin_bit_cast(unsigned, f);
    u += 0x7FFFu + ((u >> 16) & 1u);
    return (unsigned short)(u >> 16);
}

union Frag { uint4 u[2]; v16bf v; };

// A fragment (16x32 bf16 MxK): lane holds row m = lane&15.
// lanes 0-15: K = 0..7 / 16..23 ; lanes 16-31: K = 8..15 / 24..31.
static __device__ __forceinline__ v16bf frag_a(const unsigned short* base, int off0) {
    Frag f;
    f.u[0] = *reinterpret_cast<const uint4*>(base + off0);
    f.u[1] = *reinterpret_cast<const uint4*>(base + off0 + 16);
    return f.v;
}

// Opaque zero: loop-variant as far as the compiler knows, so weight-load
// addresses cannot be hoisted out of the t-loop (and spilled), but pointer
// provenance (global address space) is preserved.
static __device__ __forceinline__ unsigned launder_zero() {
    unsigned z = 0;
    asm volatile("" : "+v"(z));
    return z;
}

// GEMM for a 16-row block: NTT 16x16 output tiles per wave, K = KB*32.
// A from LDS, B from packed weights at immediate offsets; B fragments
// ping-pong double-buffered so loads stay a full kb ahead of the WMMAs.
template <int KB, int NTT>
static __device__ __forceinline__ void gemm_rows16(
    const unsigned short* __restrict__ Abase, int aoff,
    const char* __restrict__ pB, v8f* acc) {
    Frag buf[2][NTT];
    #pragma unroll
    for (int ti = 0; ti < NTT; ++ti) {
        const uint4* q = reinterpret_cast<const uint4*>(pB + (size_t)(ti * KB) * 1024);
        buf[0][ti].u[0] = q[0];
        buf[0][ti].u[1] = q[1];
    }
    #pragma unroll
    for (int kb = 0; kb < KB; ++kb) {
        const int cb = kb & 1, nb = cb ^ 1;
        if (kb + 1 < KB) {
            #pragma unroll
            for (int ti = 0; ti < NTT; ++ti) {
                const uint4* q = reinterpret_cast<const uint4*>(
                    pB + (size_t)(ti * KB + kb + 1) * 1024);
                buf[nb][ti].u[0] = q[0];
                buf[nb][ti].u[1] = q[1];
            }
        }
        v16bf a = frag_a(Abase, aoff + kb * 32);
        #pragma unroll
        for (int ti = 0; ti < NTT; ++ti)
            acc[ti] = __builtin_amdgcn_wmma_f32_16x16x32_bf16(
                false, a, false, buf[cb][ti].v, (short)0, acc[ti], false, false);
    }
}

// Pack row-major f32 W[K][N] into per-lane bf16 B-fragment layout:
// dst[((nt*KB + kb)*32 + lane)*16 + e] = W[kb*32 + (lane>>4)*8 + (e&7) + (e>>3)*16][nt*16 + (lane&15)]
__global__ void pack_w_kernel(const float* __restrict__ src,
                              unsigned short* __restrict__ dst, int K, int N) {
    int tid = blockIdx.x * 256 + threadIdx.x;
    if (tid >= K * N) return;
    int fi   = tid >> 9;
    int rem  = tid & 511;
    int lane = rem >> 4;
    int e    = rem & 15;
    int KB   = K >> 5;
    int nt   = fi / KB;
    int kb   = fi - nt * KB;
    int k    = kb * 32 + ((lane >> 4) << 3) + (e & 7) + ((e >> 3) << 4);
    int n    = nt * 16 + (lane & 15);
    dst[tid] = f2bf(src[(size_t)k * N + n]);
}

// ---------------------------------------------------------------------------
// Persistent scan + decode. grid = 64 blocks x 512 threads.
// LDS Xs = [feat_t | hs] (bf16); hs columns live in LDS for all 512 steps.
// ---------------------------------------------------------------------------
__global__ __launch_bounds__(NTHREADS)
void rnn_scan_kernel(const float* __restrict__ feats, const int* __restrict__ lengths,
                     const float* __restrict__ h0,
                     const float* __restrict__ b1, const float* __restrict__ b2,
                     const float* __restrict__ bd1, const float* __restrict__ bd2,
                     const unsigned short* __restrict__ W1p,
                     const unsigned short* __restrict__ W2p,
                     const unsigned short* __restrict__ Wd1p,
                     const unsigned short* __restrict__ Wd2p,
                     float* __restrict__ out) {
    __shared__ __align__(16) unsigned short Xs[ROWS * XSTRIDE];
    __shared__ __align__(16) unsigned short Hs[ROWS * HSTRIDE];
    __shared__ int lenS[ROWS];

    const int tid  = threadIdx.x;
    const int row0 = blockIdx.x * ROWS;

    // ---- one-time init: lengths + hs columns of Xs from h0 ----------------
    if (tid < ROWS) lenS[tid] = lengths[row0 + tid];
    for (int c = tid; c < H_; c += NTHREADS) {
        unsigned short hv = f2bf(h0[c]);
        #pragma unroll
        for (int r = 0; r < ROWS; ++r) Xs[r * XSTRIDE + F_ + c] = hv;
    }

    const int wave = tid >> 5, lane = tid & 31;
    const int hi = lane >> 4, m = lane & 15;
    const int nt0 = wave * NT;
    const int aoff  = m * XSTRIDE + hi * 8;        // layer-1 A (all 768 cols)
    const int aoffD = aoff + F_;                   // decode-1 A (hs cols only)
    const int hoff  = m * HSTRIDE + hi * 8;        // layer-2 A

    const int KB1 = XK / 32;  // 24
    const int KB2 = H_ / 32;  // 16
    const char* pB1 = (const char*)W1p
        + ((size_t)nt0 * KB1 * 512 + (size_t)lane * 16) * 2;
    const char* pB2 = (const char*)W2p
        + ((size_t)nt0 * KB2 * 512 + (size_t)lane * 16) * 2;

    // per-thread feature slice: row fr_r, 8 consecutive cols at fr_c
    const int fr_r = tid >> 5;           // 0..15
    const int fr_c = (tid & 31) * 8;     // 0..248
    const float* frow = feats + ((size_t)(row0 + fr_r) * T_) * F_ + fr_c;
    float4 fA = *reinterpret_cast<const float4*>(frow);
    float4 fB = *reinterpret_cast<const float4*>(frow + 4);

    float bias1[NT], bias2[NT];
    #pragma unroll
    for (int ti = 0; ti < NT; ++ti) {
        bias1[ti] = b1[(nt0 + ti) * 16 + m];
        bias2[ti] = b2[(nt0 + ti) * 16 + m];
    }

    // ======================= the 512-step scan =============================
    #pragma unroll 1
    for (int t = 0; t < T_; ++t) {
        // opaque zero offset: keeps weight loads inside the iteration while
        // preserving the global address space (global_load_b128, LOADcnt only)
        const unsigned z = launder_zero();
        const char* w1 = pB1 + z;
        const char* w2 = pB2 + z;

        // stage feat_t (preloaded in regs) into Xs, prefetch feat_{t+1}
        uint4 pk;
        pk.x = (unsigned)f2bf(fA.x) | ((unsigned)f2bf(fA.y) << 16);
        pk.y = (unsigned)f2bf(fA.z) | ((unsigned)f2bf(fA.w) << 16);
        pk.z = (unsigned)f2bf(fB.x) | ((unsigned)f2bf(fB.y) << 16);
        pk.w = (unsigned)f2bf(fB.z) | ((unsigned)f2bf(fB.w) << 16);
        *reinterpret_cast<uint4*>(&Xs[fr_r * XSTRIDE + fr_c]) = pk;
        if (t + 1 < T_) {
            const float* p = frow + (size_t)(t + 1) * F_;
            fA = *reinterpret_cast<const float4*>(p);
            fB = *reinterpret_cast<const float4*>(p + 4);
        }
        __syncthreads();   // X_t complete (feat cols + hs cols)

        // ---- layer 1: [16x768]@[768x512] + b1, relu -> Hs -----------------
        v8f acc[NT];
        #pragma unroll
        for (int ti = 0; ti < NT; ++ti)
            #pragma unroll
            for (int j = 0; j < 8; ++j) acc[ti][j] = bias1[ti];
        gemm_rows16<KB1, NT>(Xs, aoff, w1, acc);
        #pragma unroll
        for (int ti = 0; ti < NT; ++ti) {
            int n = (nt0 + ti) * 16 + m;
            #pragma unroll
            for (int j = 0; j < 8; ++j) {
                float v = acc[ti][j];
                v = v > 0.f ? v : 0.f;
                Hs[(j + 8 * hi) * HSTRIDE + n] = f2bf(v);
            }
        }
        __syncthreads();   // Hs complete; all Xs reads done

        // ---- layer 2: [16x512]@[512x512] + b2, relu, masked hs <- Xs ------
        v8f acc2[NT];
        #pragma unroll
        for (int ti = 0; ti < NT; ++ti)
            #pragma unroll
            for (int j = 0; j < 8; ++j) acc2[ti][j] = bias2[ti];
        gemm_rows16<KB2, NT>(Hs, hoff, w2, acc2);
        #pragma unroll
        for (int ti = 0; ti < NT; ++ti) {
            int n = (nt0 + ti) * 16 + m;
            #pragma unroll
            for (int j = 0; j < 8; ++j) {
                int mm = j + 8 * hi;
                if (t < lenS[mm]) {          // inactive rows keep old hs
                    float v = acc2[ti][j];
                    Xs[mm * XSTRIDE + F_ + n] = f2bf(v > 0.f ? v : 0.f);
                }
            }
        }
        // next iteration's top barrier orders these writes vs. layer-1 reads
    }
    __syncthreads();

    // ======================= decode head ===================================
    // decode1: relu(hs @ Wd1 + bd1) -> Hs
    {
        const char* pD1 = (const char*)Wd1p
            + ((size_t)nt0 * KB2 * 512 + (size_t)lane * 16) * 2;
        v8f acc[NT];
        #pragma unroll
        for (int ti = 0; ti < NT; ++ti) {
            float bv = bd1[(nt0 + ti) * 16 + m];
            #pragma unroll
            for (int j = 0; j < 8; ++j) acc[ti][j] = bv;
        }
        gemm_rows16<KB2, NT>(Xs, aoffD, pD1, acc);
        #pragma unroll
        for (int ti = 0; ti < NT; ++ti) {
            int n = (nt0 + ti) * 16 + m;
            #pragma unroll
            for (int j = 0; j < 8; ++j) {
                float v = acc[ti][j];
                v = v > 0.f ? v : 0.f;
                Hs[(j + 8 * hi) * HSTRIDE + n] = f2bf(v);
            }
        }
    }
    __syncthreads();

    // decode2: sigmoid(Hs @ Wd2 + bd2), N=64 -> 4 tiles on waves 0..3
    if (wave < 4) {
        const char* pD2 = (const char*)Wd2p
            + ((size_t)wave * KB2 * 512 + (size_t)lane * 16) * 2;
        v8f a2[1];
        float bv = bd2[wave * 16 + m];
        #pragma unroll
        for (int j = 0; j < 8; ++j) a2[0][j] = bv;
        gemm_rows16<KB2, 1>(Hs, hoff, pD2, a2);
        int n = wave * 16 + m;
        #pragma unroll
        for (int j = 0; j < 8; ++j) {
            int mm = j + 8 * hi;
            float v = a2[0][j];
            out[(size_t)(row0 + mm) * O_ + n] = 1.f / (1.f + __expf(-v));
        }
    }
}

// ---------------------------------------------------------------------------
extern "C" void kernel_launch(void* const* d_in, const int* in_sizes, int n_in,
                              void* d_out, int out_size, void* d_ws, size_t ws_size,
                              hipStream_t stream) {
    const float* feats   = (const float*)d_in[0];
    const int*   lengths = (const int*)  d_in[1];
    const float* h0      = (const float*)d_in[2];
    const float* W1      = (const float*)d_in[3];
    const float* b1      = (const float*)d_in[4];
    const float* W2      = (const float*)d_in[5];
    const float* b2      = (const float*)d_in[6];
    const float* Wd1     = (const float*)d_in[7];
    const float* bd1     = (const float*)d_in[8];
    const float* Wd2     = (const float*)d_in[9];
    const float* bd2     = (const float*)d_in[10];
    float* out = (float*)d_out;

    // workspace: packed bf16 weights (~1.9 MB)
    char* ws = (char*)d_ws;
    unsigned short* W1p  = (unsigned short*)(ws);
    unsigned short* W2p  = (unsigned short*)(ws + (size_t)XK * H_ * 2);
    unsigned short* Wd1p = (unsigned short*)(ws + ((size_t)XK * H_ + (size_t)H_ * H_) * 2);
    unsigned short* Wd2p = (unsigned short*)(ws + ((size_t)XK * H_ + 2ull * H_ * H_) * 2);

    pack_w_kernel<<<(XK * H_ + 255) / 256, 256, 0, stream>>>(W1,  W1p,  XK, H_);
    pack_w_kernel<<<(H_ * H_ + 255) / 256, 256, 0, stream>>>(W2,  W2p,  H_, H_);
    pack_w_kernel<<<(H_ * H_ + 255) / 256, 256, 0, stream>>>(Wd1, Wd1p, H_, H_);
    pack_w_kernel<<<(H_ * O_ + 255) / 256, 256, 0, stream>>>(Wd2, Wd2p, H_, O_);

    rnn_scan_kernel<<<B_ / ROWS, NTHREADS, 0, stream>>>(
        feats, lengths, h0, b1, b2, bd1, bd2, W1p, W2p, Wd1p, Wd2p, out);
}
